// FusedExpertsNetwork_86543591014695
// MI455X (gfx1250) — matrix operand order repaired and checked by reading, not compiled, not run on hardware
//
#include <hip/hip_runtime.h>
#include <hip/hip_bf16.h>

typedef __attribute__((ext_vector_type(16))) __bf16 v16bf;
typedef __attribute__((ext_vector_type(8)))  __bf16 v8bf;
typedef __attribute__((ext_vector_type(4)))  __bf16 v4bf;
typedef __attribute__((ext_vector_type(8)))  float  v8f;
typedef __attribute__((ext_vector_type(4)))  int    v4i;

#define T_TOK 8192
#define DIM   1024
#define HID   4096
#define NE    8
#define CAP   2048
#define LDT   40            // LDS row stride in halves (32 data + 8 pad)
#define LDS_TILE (128 * LDT)

#define ASG __attribute__((address_space(1)))
#define ASL __attribute__((address_space(3)))

typedef ASL __bf16* lds_bf16_p;

#if __has_builtin(__builtin_amdgcn_global_load_async_to_lds_b128)
#define USE_ASYNC 1
#else
#define USE_ASYNC 0
#endif

#if __has_builtin(__builtin_amdgcn_s_wait_asynccnt)
#define WAIT_ASYNC(n) __builtin_amdgcn_s_wait_asynccnt(n)
#else
#define WAIT_ASYNC(n) asm volatile("s_wait_asynccnt %0" :: "i"(n) : "memory")
#endif

// ---------------------------------------------------------------- routing ---
// One block, 256 threads (8 waves). Token-ordered slot assignment so the
// capacity-drop set matches the reference cumsum semantics exactly.
__global__ void route_kernel(const float* __restrict__ logits,
                             int* __restrict__ eidx, float* __restrict__ gate,
                             int* __restrict__ slot, int* __restrict__ tok_of) {
    __shared__ int counts[NE];
    __shared__ int wave_cnt[8][NE];
    __shared__ int wave_off[8][NE];
    const int tid = threadIdx.x;
    const int lane = tid & 31;
    const int wv = tid >> 5;
    if (tid < NE) counts[tid] = 0;
    __syncthreads();
    for (int base = 0; base < T_TOK; base += 256) {
        const int t = base + tid;
        float l[NE];
        float m = -3.0e38f;
        int am = 0;
        #pragma unroll
        for (int e = 0; e < NE; ++e) {
            l[e] = logits[t * NE + e];
            if (l[e] > m) { m = l[e]; am = e; }
        }
        float s = 0.f;
        #pragma unroll
        for (int e = 0; e < NE; ++e) s += __expf(l[e] - m);
        eidx[t] = am;
        gate[t] = 1.0f / s;   // exp(m-m)/sum
        // ordered rank within wave, per expert
        int myrank = 0;
        #pragma unroll
        for (int e = 0; e < NE; ++e) {
            unsigned long long b = __ballot(am == e);
            if (e == am) myrank = __popcll(b & ((1ull << lane) - 1ull));
            if (lane == 0) wave_cnt[wv][e] = (int)__popcll(b);
        }
        __syncthreads();
        if (wv == 0 && lane < NE) {
            const int e = lane;
            int acc = counts[e];
            #pragma unroll
            for (int w = 0; w < 8; ++w) { wave_off[w][e] = acc; acc += wave_cnt[w][e]; }
            counts[e] = acc;
        }
        __syncthreads();
        const int myslot = wave_off[wv][am] + myrank;
        slot[t] = myslot;
        if (myslot < CAP) tok_of[am * CAP + myslot] = t;
        __syncthreads();
    }
}

// --------------------------------------------------------------- dispatch ---
// grid = T tokens, 256 threads, float4 per thread (D = 1024).
__global__ void dispatch_kernel(const float* __restrict__ x,
                                const int* __restrict__ eidx,
                                const int* __restrict__ slot,
                                __bf16* __restrict__ disp,
                                float* __restrict__ out) {
    const int t = blockIdx.x;
    const int tid = threadIdx.x;
    const int s = slot[t];
    if (s < CAP) {
        const int e = eidx[t];
        float4 v = ((const float4*)(x + (size_t)t * DIM))[tid];
        v4bf o = { (__bf16)v.x, (__bf16)v.y, (__bf16)v.z, (__bf16)v.w };
        ((v4bf*)(disp + ((size_t)e * CAP + s) * DIM))[tid] = o;
    } else {
        float4 z = { 0.f, 0.f, 0.f, 0.f };
        ((float4*)(out + (size_t)t * DIM))[tid] = z;
    }
}

// ------------------------------------------------------------ weight prep ---
__global__ void cvt_w1_kernel(const float* __restrict__ w, __bf16* __restrict__ wb,
                              long long n4) {
    long long i = (long long)blockIdx.x * blockDim.x + threadIdx.x;
    if (i < n4) {
        float4 v = ((const float4*)w)[i];
        v4bf o = { (__bf16)v.x, (__bf16)v.y, (__bf16)v.z, (__bf16)v.w };
        ((v4bf*)wb)[i] = o;
    }
}

// fc2_w (E,H,D) fp32  ->  w2t (E,D,H) bf16  (K = H contiguous for GEMM2)
__global__ void w2_transpose_kernel(const float* __restrict__ w2,
                                    __bf16* __restrict__ w2t) {
    __shared__ float tile[32][33];
    const int e = blockIdx.z;
    const int h0 = blockIdx.x * 32;
    const int o0 = blockIdx.y * 32;
    const float* src = w2 + (size_t)e * HID * DIM;
    __bf16* dst = w2t + (size_t)e * DIM * HID;
    const int tx = threadIdx.x & 31;
    const int ty = threadIdx.x >> 5;   // 0..7
    #pragma unroll
    for (int i = 0; i < 32; i += 8)
        tile[ty + i][tx] = src[(size_t)(h0 + ty + i) * DIM + o0 + tx];
    __syncthreads();
    #pragma unroll
    for (int i = 0; i < 32; i += 8)
        dst[(size_t)(o0 + ty + i) * HID + h0 + tx] = (__bf16)tile[tx][ty + i];
}

// ------------------------------------------------------------- GEMM core ----
__device__ __forceinline__ v16bf cat16(v8bf a, v8bf b) {
    v16bf r;
    #pragma unroll
    for (int i = 0; i < 8; ++i) { r[i] = a[i]; r[i + 8] = b[i]; }
    return r;
}

// Issue one 32-deep K-tile (A: 128 rows, B: 128 rows) into LDS buffers.
// 4 x 16B ops per thread (2 for A, 2 for B).
__device__ __forceinline__ void tile_copy_issue(const __bf16* __restrict__ gA,
                                                const __bf16* __restrict__ gB,
                                                int K, int m0, int n0, int k0,
                                                lds_bf16_p As, lds_bf16_p Bs, int tid) {
    #pragma unroll
    for (int i = 0; i < 2; ++i) {
        const int c = tid + i * 256;          // 0..511
        const int row = c >> 2;               // 0..127
        const int cc = (c & 3) * 8;           // 0,8,16,24 halves
#if USE_ASYNC
        __builtin_amdgcn_global_load_async_to_lds_b128(
            (ASG v4i*)&gA[(size_t)(m0 + row) * K + k0 + cc],
            (ASL v4i*)&As[row * LDT + cc], 0, 0);
        __builtin_amdgcn_global_load_async_to_lds_b128(
            (ASG v4i*)&gB[(size_t)(n0 + row) * K + k0 + cc],
            (ASL v4i*)&Bs[row * LDT + cc], 0, 0);
#else
        *(ASL v8bf*)&As[row * LDT + cc] =
            *(const v8bf*)&gA[(size_t)(m0 + row) * K + k0 + cc];
        *(ASL v8bf*)&Bs[row * LDT + cc] =
            *(const v8bf*)&gB[(size_t)(n0 + row) * K + k0 + cc];
#endif
    }
}

// 8 WMMAs on one staged 32-deep K-tile (wave tile 64x32).
__device__ __forceinline__ void compute_tile(const ASL __bf16* As, const ASL __bf16* Bs,
                                             int waveM, int waveN, int lr, int hs,
                                             v8f acc[4][2]) {
    v16bf bfrag[2];
    #pragma unroll
    for (int ni = 0; ni < 2; ++ni) {
        const int n = waveN * 32 + ni * 16 + lr;
        v8bf lo = *(const ASL v8bf*)&Bs[n * LDT + hs * 16];
        v8bf hi = *(const ASL v8bf*)&Bs[n * LDT + hs * 16 + 8];
        bfrag[ni] = cat16(lo, hi);
    }
    #pragma unroll
    for (int mi = 0; mi < 4; ++mi) {
        const int m = waveM * 64 + mi * 16 + lr;
        v8bf lo = *(const ASL v8bf*)&As[m * LDT + hs * 16];
        v8bf hi = *(const ASL v8bf*)&As[m * LDT + hs * 16 + 8];
        v16bf afrag = cat16(lo, hi);
        #pragma unroll
        for (int ni = 0; ni < 2; ++ni)
            acc[mi][ni] = __builtin_amdgcn_wmma_f32_16x16x32_bf16(
                false, afrag, false, bfrag[ni], (short)0, acc[mi][ni],
                false, false);
    }
}

// C[m0..+128, n0..+128] += A[M,K] * B[N,K]^T, both row-major, K-contiguous.
// 256 threads = 8 waves (2 x 4), wave tile 64x32, K-step 32, double-buffered,
// software-pipelined with async-to-LDS staging.
__device__ __forceinline__ void gemm_mainloop(const __bf16* __restrict__ A,
                                              const __bf16* __restrict__ B,
                                              int K, int m0, int n0,
                                              lds_bf16_p lds, v8f acc[4][2]) {
    const int tid = threadIdx.x;
    const int lane = tid & 31;
    const int wv = tid >> 5;
    const int waveM = wv >> 2;     // 0..1
    const int waveN = wv & 3;      // 0..3
    const int lr = lane & 15;
    const int hs = lane >> 4;      // 0..1

    // buffer p occupies [p*2*LDS_TILE, p*2*LDS_TILE + 2*LDS_TILE)
    // prologue: stage tile 0 into buffer 0
    tile_copy_issue(A, B, K, m0, n0, 0, lds, lds + LDS_TILE, tid);

    int p = 0;
    int k0 = 0;
    for (; k0 < K - 32; k0 += 32) {
        const lds_bf16_p cur = lds + p * (2 * LDS_TILE);
        const lds_bf16_p nxt = lds + (p ^ 1) * (2 * LDS_TILE);
        tile_copy_issue(A, B, K, m0, n0, k0 + 32, nxt, nxt + LDS_TILE, tid);
#if USE_ASYNC
        WAIT_ASYNC(4);             // current tile's 4 ops are the oldest
#endif
        __syncthreads();           // current tile visible to all waves
        compute_tile(cur, cur + LDS_TILE, waveM, waveN, lr, hs, acc);
        __syncthreads();           // all reads of cur done before it is reused
        p ^= 1;
    }
    // peeled last tile
    {
        const lds_bf16_p cur = lds + p * (2 * LDS_TILE);
#if USE_ASYNC
        WAIT_ASYNC(0);
#endif
        __syncthreads();
        compute_tile(cur, cur + LDS_TILE, waveM, waveN, lr, hs, acc);
    }
}

// GEMM1: h = relu(disp * fc1_w^T + b1)  -> bf16
__global__ void __launch_bounds__(256)
gemm1_kernel(const __bf16* __restrict__ disp, const __bf16* __restrict__ w1b,
             const float* __restrict__ b1, __bf16* __restrict__ hbuf) {
    __shared__ __align__(16) __bf16 lds[4 * LDS_TILE];
    const int e = blockIdx.z;
    const int n0 = blockIdx.x * 128;
    const int m0 = blockIdx.y * 128;
    const __bf16* A = disp + (size_t)e * CAP * DIM;
    const __bf16* B = w1b + (size_t)e * HID * DIM;
    v8f acc[4][2];
    #pragma unroll
    for (int i = 0; i < 4; ++i)
        #pragma unroll
        for (int j = 0; j < 2; ++j)
            acc[i][j] = (v8f){0.f,0.f,0.f,0.f,0.f,0.f,0.f,0.f};

    gemm_mainloop(A, B, DIM, m0, n0, (lds_bf16_p)lds, acc);

    const int tid = threadIdx.x, lane = tid & 31, wv = tid >> 5;
    const int waveM = wv >> 2, waveN = wv & 3, lr = lane & 15, hs = lane >> 4;
    #pragma unroll
    for (int mi = 0; mi < 4; ++mi) {
        #pragma unroll
        for (int ni = 0; ni < 2; ++ni) {
            const int col = n0 + waveN * 32 + ni * 16 + lr;
            const float bias = b1[(size_t)e * HID + col];
            #pragma unroll
            for (int r = 0; r < 8; ++r) {
                const int row = m0 + waveM * 64 + mi * 16 + hs * 8 + r;
                float v = acc[mi][ni][r] + bias;
                v = v > 0.f ? v : 0.f;
                hbuf[((size_t)e * CAP + row) * HID + col] = (__bf16)v;
            }
        }
    }
}

// GEMM2: y = h * fc2_w + b2, scaled by gate, scattered to out via tok_of.
__global__ void __launch_bounds__(256)
gemm2_kernel(const __bf16* __restrict__ hbuf, const __bf16* __restrict__ w2t,
             const float* __restrict__ b2, const int* __restrict__ tok_of,
             const float* __restrict__ gate, float* __restrict__ out) {
    __shared__ __align__(16) __bf16 lds[4 * LDS_TILE];
    __shared__ int   s_tok[128];
    __shared__ float s_gate[128];
    const int e = blockIdx.z;
    const int n0 = blockIdx.x * 128;
    const int m0 = blockIdx.y * 128;
    if (threadIdx.x < 128) {
        const int tk = tok_of[e * CAP + m0 + threadIdx.x];
        s_tok[threadIdx.x] = tk;
        s_gate[threadIdx.x] = (tk >= 0) ? gate[tk] : 0.f;
    }
    const __bf16* A = hbuf + (size_t)e * CAP * HID;
    const __bf16* B = w2t + (size_t)e * DIM * HID;
    v8f acc[4][2];
    #pragma unroll
    for (int i = 0; i < 4; ++i)
        #pragma unroll
        for (int j = 0; j < 2; ++j)
            acc[i][j] = (v8f){0.f,0.f,0.f,0.f,0.f,0.f,0.f,0.f};

    gemm_mainloop(A, B, HID, m0, n0, (lds_bf16_p)lds, acc);

    const int tid = threadIdx.x, lane = tid & 31, wv = tid >> 5;
    const int waveM = wv >> 2, waveN = wv & 3, lr = lane & 15, hs = lane >> 4;
    #pragma unroll
    for (int mi = 0; mi < 4; ++mi) {
        #pragma unroll
        for (int ni = 0; ni < 2; ++ni) {
            const int col = n0 + waveN * 32 + ni * 16 + lr;
            const float bias = b2[(size_t)e * DIM + col];
            #pragma unroll
            for (int r = 0; r < 8; ++r) {
                const int rloc = waveM * 64 + mi * 16 + hs * 8 + r;
                const int tk = s_tok[rloc];
                if (tk >= 0) {
                    out[(size_t)tk * DIM + col] =
                        (acc[mi][ni][r] + bias) * s_gate[rloc];
                }
            }
        }
    }
}

// ------------------------------------------------------------------ launch --
extern "C" void kernel_launch(void* const* d_in, const int* in_sizes, int n_in,
                              void* d_out, int out_size, void* d_ws, size_t ws_size,
                              hipStream_t stream) {
    const float* x      = (const float*)d_in[0];
    const float* logits = (const float*)d_in[1];
    const float* w1     = (const float*)d_in[2];
    const float* w2     = (const float*)d_in[3];
    const float* b1     = (const float*)d_in[4];
    const float* b2     = (const float*)d_in[5];
    float* out = (float*)d_out;

    char* ws = (char*)d_ws;
    const size_t SZ_W1B  = (size_t)NE * HID * DIM * 2;   // 64 MB
    const size_t SZ_W2T  = (size_t)NE * DIM * HID * 2;   // 64 MB
    const size_t SZ_DISP = (size_t)NE * CAP * DIM * 2;   // 32 MB
    const size_t SZ_H    = (size_t)NE * CAP * HID * 2;   // 128 MB
    __bf16* w1b  = (__bf16*)(ws);
    __bf16* w2t  = (__bf16*)(ws + SZ_W1B);
    __bf16* disp = (__bf16*)(ws + SZ_W1B + SZ_W2T);
    __bf16* hbuf = (__bf16*)(ws + SZ_W1B + SZ_W2T + SZ_DISP);
    char* tail   = ws + SZ_W1B + SZ_W2T + SZ_DISP + SZ_H;
    int*   eidx   = (int*)(tail);
    float* gatep  = (float*)(tail + 32768);
    int*   slot   = (int*)(tail + 65536);
    int*   tok_of = (int*)(tail + 98304);

    (void)hipMemsetAsync(tok_of, 0xFF, (size_t)NE * CAP * sizeof(int), stream);

    route_kernel<<<1, 256, 0, stream>>>(logits, eidx, gatep, slot, tok_of);
    dispatch_kernel<<<T_TOK, 256, 0, stream>>>(x, eidx, slot, disp, out);

    const long long n4 = (long long)NE * HID * DIM / 4;
    cvt_w1_kernel<<<(unsigned)((n4 + 255) / 256), 256, 0, stream>>>(w1, w1b, n4);
    w2_transpose_kernel<<<dim3(HID / 32, DIM / 32, NE), 256, 0, stream>>>(w2, w2t);

    gemm1_kernel<<<dim3(HID / 128, CAP / 128, NE), 256, 0, stream>>>(disp, w1b, b1, hbuf);
    gemm2_kernel<<<dim3(DIM / 128, CAP / 128, NE), 256, 0, stream>>>(hbuf, w2t, b2, tok_of, gatep, out);
}
